// TransformersBasedFormulaEmbeddingLayer_75831942578501
// MI455X (gfx1250) — matrix operand order repaired and checked by reading, not compiled, not run on hardware
//
#include <hip/hip_runtime.h>

// ---------------------------------------------------------------------------
// CDNA5 / gfx1250 implementation of the transformers-based formula embedding
// layer.  GEMMs run on v_wmma_f32_16x16x32_bf16; scatter-mean uses hardware
// f32 global atomics; counts are idx-only and computed once per launch.
// M-tile = 32 groundings: each B fragment feeds two WMMAs.
// ---------------------------------------------------------------------------

typedef __attribute__((ext_vector_type(16))) __bf16 v16bf;
typedef __attribute__((ext_vector_type(8)))  float  v8f;

#define EMB 128

__device__ __forceinline__ unsigned short f32_to_bf16(float f) {
    unsigned int u = __float_as_uint(f);
    unsigned int r = 0x7FFFu + ((u >> 16) & 1u);   // round-to-nearest-even
    return (unsigned short)((u + r) >> 16);
}

// -------------------------------- converts --------------------------------

__global__ __launch_bounds__(256) void convert_x_kernel(
    const float* __restrict__ x, unsigned short* __restrict__ outBf, int n) {
    int i = blockIdx.x * blockDim.x + threadIdx.x;
    if (i < n) outBf[i] = f32_to_bf16(x[i]);
}

// W [T, K, K] row-major f32  ->  WT [T, K, K] column-major bf16 (WT[t][j][k])
__global__ __launch_bounds__(256) void convert_w_kernel(
    const float* __restrict__ W, unsigned short* __restrict__ WT,
    int T, int K) {
    int total = T * K * K;
    for (int i = blockIdx.x * blockDim.x + threadIdx.x; i < total;
         i += gridDim.x * blockDim.x) {
        int t = i / (K * K);
        int rem = i - t * K * K;
        int k = rem / K;
        int j = rem - k * K;
        WT[(size_t)t * K * K + (size_t)j * K + k] = f32_to_bf16(W[i]);
    }
}

// ------------------------------- counts -----------------------------------

__global__ __launch_bounds__(256) void count_kernel(
    const int* __restrict__ idxFlat, int n, float* __restrict__ cnt) {
    int i = blockIdx.x * blockDim.x + threadIdx.x;
    if (i < n) unsafeAtomicAdd(&cnt[idxFlat[i]], 1.0f);
}

// --------------------------- formula pass (WMMA) ---------------------------
// Block: 256 threads (8 waves).  M-tile = 32 groundings (two 16-row WMMA
// slabs).  Wave w owns output columns [(w*ATOMS)*16, (w*ATOMS+ATOMS)*16):
// exactly ATOMS 16-wide D-tiles per slab.  K = ATOMS*128, stepped 32/iter.
// Each B fragment is reused by both M slabs -> 2 WMMAs per B fetch.

template <int ATOMS>
__global__ __launch_bounds__(256) void formula_pass_kernel(
    const unsigned short* __restrict__ atomsBf,   // [N,128] bf16
    const int*            __restrict__ idx,       // [G, ATOMS]
    const unsigned short* __restrict__ WT,        // [K,K] bf16, column-major
    const float*          __restrict__ bias,      // [K]
    float*                __restrict__ sums,      // [N,128] f32 (atomic dst)
    int G) {
    constexpr int K   = ATOMS * EMB;
    constexpr int LDA = K + 16;                   // pad: kill LDS bank conflicts
    constexpr int MT  = 32;                       // groundings per block

    __shared__ unsigned short smA[MT * LDA];      // gathered A tile, bf16 bits
    __shared__ int            smIdx[MT * ATOMS];

    const int tid   = threadIdx.x;
    const int lane  = tid & 31;
    const int wave  = tid >> 5;
    const int gBase = blockIdx.x * MT;

    // ---- stage idx tile (tail-guarded) ----
    for (int t = tid; t < MT * ATOMS; t += 256) {
        int gi = gBase * ATOMS + t;
        smIdx[t] = (gi < G * ATOMS) ? idx[gi] : 0;
    }
    __syncthreads();

    // ---- gather MT rows x K bf16 into LDS (16B vectors) ----
    constexpr int VECS = MT * K / 8;              // 1024 (A=2) / 1536 (A=3)
    for (int v = tid; v < VECS; v += 256) {
        int row   = v / (K / 8);
        int chunk = v - row * (K / 8);
        int a     = chunk >> 4;                   // 16 8-elem chunks per atom
        int node  = smIdx[row * ATOMS + a];
        const uint4* src =
            (const uint4*)(atomsBf + (size_t)node * EMB + (chunk & 15) * 8);
        *(uint4*)(&smA[row * LDA + chunk * 8]) = *src;
    }
    __syncthreads();

    const int rowA    = lane & 15;                // A-matrix row for this lane
    const int khalf   = lane >> 4;                // lane-half K selector
    const int colLane = lane & 15;                // D/B column within tile

    v8f acc[2 * ATOMS] = {};

    for (int kk = 0; kk < K; kk += 32) {
        // A fragments per ISA 16-bit A layout: halves 0..7 = K kk+khalf*8..+7,
        // halves 8..15 = K kk+16+khalf*8..+7.  Two slabs: rows 0-15, 16-31.
        union { uint4 u[2]; v16bf v; } afrag0, afrag1;
        const unsigned short* arow0 = &smA[rowA * LDA + kk + khalf * 8];
        const unsigned short* arow1 = arow0 + 16 * LDA;
        afrag0.u[0] = *(const uint4*)(arow0);
        afrag0.u[1] = *(const uint4*)(arow0 + 16);
        afrag1.u[0] = *(const uint4*)(arow1);
        afrag1.u[1] = *(const uint4*)(arow1 + 16);

#pragma unroll
        for (int nt = 0; nt < ATOMS; nt++) {
            int col = (wave * ATOMS + nt) * 16 + colLane;
            // B fragment: lane holds one column; halves 0..7 = K kk+khalf*8..+7, 8..15 = K kk+16+khalf*8..+7
            union { uint4 u[2]; v16bf v; } bfrag;
            const unsigned short* bcol =
                WT + (size_t)col * K + kk + khalf * 8;
            bfrag.u[0] = *(const uint4*)(bcol);
            bfrag.u[1] = *(const uint4*)(bcol + 16);

            acc[2 * nt + 0] = __builtin_amdgcn_wmma_f32_16x16x32_bf16(
                false, afrag0.v, false, bfrag.v, (short)0, acc[2 * nt + 0],
                false, false);
            acc[2 * nt + 1] = __builtin_amdgcn_wmma_f32_16x16x32_bf16(
                false, afrag1.v, false, bfrag.v, (short)0, acc[2 * nt + 1],
                false, false);
        }
    }

    // ---- epilogue: bias, ReLU, scatter-add into per-node sums ----
#pragma unroll
    for (int nt = 0; nt < ATOMS; nt++) {
        int col   = (wave * ATOMS + nt) * 16 + colLane;
        int aAtom = col >> 7;                     // which atom slot (E=128)
        int e     = col & 127;
        float bv  = bias[col];
#pragma unroll
        for (int slab = 0; slab < 2; slab++) {
#pragma unroll
            for (int r = 0; r < 8; r++) {
                int m = slab * 16 + r + 8 * khalf;  // D layout: VGPR r -> row
                if (gBase + m < G) {
                    int   node = smIdx[m * ATOMS + aAtom];
                    float v    = acc[2 * nt + slab][r] + bv;
                    v = v > 0.0f ? v : 0.0f;
                    unsafeAtomicAdd(&sums[(size_t)node * EMB + e], v);
                }
            }
        }
    }
}

// ------------------------------ finalize -----------------------------------
// atoms = divide_no_nan(sums0,cnt0) + divide_no_nan(sums1,cnt1)
// writes f32 (d_out) and bf16 (next iteration's gather source)

__global__ __launch_bounds__(256) void finalize_kernel(
    const float* __restrict__ sums0, const float* __restrict__ cnt0,
    const float* __restrict__ sums1, const float* __restrict__ cnt1,
    float* __restrict__ outF, unsigned short* __restrict__ outBf, int total) {
    int i = blockIdx.x * blockDim.x + threadIdx.x;
    if (i >= total) return;
    int n = i >> 7;
    float c0 = cnt0[n], c1 = cnt1[n];
    float o0 = (c0 > 0.0f) ? sums0[i] / c0 : 0.0f;
    float o1 = (c1 > 0.0f) ? sums1[i] / c1 : 0.0f;
    float v  = o0 + o1;
    outF[i]  = v;
    outBf[i] = f32_to_bf16(v);
}

// ------------------------------ launcher -----------------------------------

static inline size_t align256(size_t x) { return (x + 255) & ~(size_t)255; }

extern "C" void kernel_launch(void* const* d_in, const int* in_sizes, int n_in,
                              void* d_out, int out_size, void* d_ws,
                              size_t ws_size, hipStream_t stream) {
    const float* x    = (const float*)d_in[0];   // [1,N,128]
    const float* W_f0 = (const float*)d_in[1];   // [T,256,256]
    const float* b_f0 = (const float*)d_in[2];   // [T,256]
    const float* W_f1 = (const float*)d_in[3];   // [T,384,384]
    const float* b_f1 = (const float*)d_in[4];   // [T,384]
    const int*   idx0 = (const int*)d_in[6];     // [G0,2]
    const int*   idx1 = (const int*)d_in[7];     // [G1,3]

    const int NE = in_sizes[0];                  // N*128
    const int N  = NE / EMB;
    const int T  = in_sizes[2] / 256;            // b_f0 is [T,256]
    const int G0 = in_sizes[6] / 2;
    const int G1 = in_sizes[7] / 3;
    const int K0 = 256, K1 = 384;

    // workspace carve-up
    char*  ws  = (char*)d_ws;
    size_t off = 0;
    float* sums0 = (float*)(ws + off);          off += align256((size_t)NE * 4);
    float* sums1 = (float*)(ws + off);          off += align256((size_t)NE * 4);
    float* cnt0  = (float*)(ws + off);          off += align256((size_t)N * 4);
    float* cnt1  = (float*)(ws + off);          off += align256((size_t)N * 4);
    unsigned short* atomsBf = (unsigned short*)(ws + off);
    off += align256((size_t)NE * 2);
    unsigned short* W0T = (unsigned short*)(ws + off);
    off += align256((size_t)T * K0 * K0 * 2);
    unsigned short* W1T = (unsigned short*)(ws + off);
    off += align256((size_t)T * K1 * K1 * 2);
    (void)ws_size;

    // zero accumulators / counts
    hipMemsetAsync(sums0, 0, (size_t)NE * 4, stream);
    hipMemsetAsync(sums1, 0, (size_t)NE * 4, stream);
    hipMemsetAsync(cnt0, 0, (size_t)N * 4, stream);
    hipMemsetAsync(cnt1, 0, (size_t)N * 4, stream);

    // bf16 conversions (x and transposed weights)
    convert_x_kernel<<<(NE + 255) / 256, 256, 0, stream>>>(x, atomsBf, NE);
    convert_w_kernel<<<1024, 256, 0, stream>>>(W_f0, W0T, T, K0);
    convert_w_kernel<<<1024, 256, 0, stream>>>(W_f1, W1T, T, K1);

    // counts depend only on idx: compute once
    count_kernel<<<(G0 * 2 + 255) / 256, 256, 0, stream>>>(idx0, G0 * 2, cnt0);
    count_kernel<<<(G1 * 3 + 255) / 256, 256, 0, stream>>>(idx1, G1 * 3, cnt1);

    const int blocks0 = (G0 + 31) / 32;
    const int blocks1 = (G1 + 31) / 32;

    for (int t = 0; t < T; ++t) {
        formula_pass_kernel<2><<<blocks0, 256, 0, stream>>>(
            atomsBf, idx0, W0T + (size_t)t * K0 * K0, b_f0 + (size_t)t * K0,
            sums0, G0);
        formula_pass_kernel<3><<<blocks1, 256, 0, stream>>>(
            atomsBf, idx1, W1T + (size_t)t * K1 * K1, b_f1 + (size_t)t * K1,
            sums1, G1);
        finalize_kernel<<<(NE + 255) / 256, 256, 0, stream>>>(
            sums0, cnt0, sums1, cnt1, (float*)d_out, atomsBf, NE);
        if (t + 1 < T) {
            hipMemsetAsync(sums0, 0, (size_t)NE * 4, stream);
            hipMemsetAsync(sums1, 0, (size_t)NE * 4, stream);
        }
    }
}
